// SlidingWindowSelfAttention_22746146799844
// MI455X (gfx1250) — matrix operand order repaired and checked by reading, not compiled, not run on hardware
//
#include <hip/hip_runtime.h>

// ---------------- problem constants ----------------
#define Bc  2
#define Tc  4096
#define Cc  1024
#define Hc  16
#define HDc 64
#define Wc  256
#define NWc (Tc / Wc)          // 16 windows
#define MTOT ((size_t)Bc * Tc) // 8192 rows

typedef __bf16 bf16_t;
typedef __attribute__((ext_vector_type(16))) __bf16 v16bf;
typedef __attribute__((ext_vector_type(8)))  __bf16 v8bf;
typedef __attribute__((ext_vector_type(8)))  float  v8f;

union AFrag { v16bf v; v8bf h[2]; };

static __device__ __forceinline__ v8f wmma_bf16(v16bf a, v16bf b, v8f c) {
    return __builtin_amdgcn_wmma_f32_16x16x32_bf16(false, a, false, b, (short)0, c,
                                                   false, false);
}

// Async copy of one 128-byte row (per lane) from global to LDS.
// CDNA5 GLOBAL_LOAD_ASYNC_TO_LDS_B128: vdst = per-lane LDS byte address,
// vaddr = 64-bit global address; INST_OFFSET is added to BOTH addresses.
// Tracked with ASYNCcnt.
#define ASYNC_LDS_ROW_128B(ldsaddr, gaddr)                                 \
    asm volatile(                                                          \
        "global_load_async_to_lds_b128 %0, %1, off\n\t"                    \
        "global_load_async_to_lds_b128 %0, %1, off offset:16\n\t"          \
        "global_load_async_to_lds_b128 %0, %1, off offset:32\n\t"          \
        "global_load_async_to_lds_b128 %0, %1, off offset:48\n\t"          \
        "global_load_async_to_lds_b128 %0, %1, off offset:64\n\t"          \
        "global_load_async_to_lds_b128 %0, %1, off offset:80\n\t"          \
        "global_load_async_to_lds_b128 %0, %1, off offset:96\n\t"          \
        "global_load_async_to_lds_b128 %0, %1, off offset:112"             \
        :: "v"(ldsaddr), "v"(gaddr) : "memory")

#define WAIT_ASYNC_0() asm volatile("s_wait_asynccnt 0x0" ::: "memory")

// LDS byte address = low 32 bits of the generic (flat) pointer
// (ISA aperture rule).
static __device__ __forceinline__ uint32_t lds_addr32(const void* p) {
    return (uint32_t)(uintptr_t)p;
}

// ---------------- f32 -> bf16 convert ----------------
__global__ __launch_bounds__(256) void cvt_f32_bf16(const float* __restrict__ in,
                                                    bf16_t* __restrict__ out,
                                                    size_t n) {
    size_t i = (size_t)blockIdx.x * blockDim.x + threadIdx.x;
    if (i < n) out[i] = (bf16_t)in[i];
}

// ---------------- generic bf16 GEMM:  out[m,n] = sum_k A[m,k] * W[n,k] --------
// A: M x K row-major bf16, W: N x K row-major bf16, out: M x N row-major f32.
// Block: 256 threads = 8 waves arranged 2(M) x 4(N); wave computes 32x64
// (2 M-tiles x 4 N-tiles = 8 WMMA per 32-wide k-step).
__global__ __launch_bounds__(256) void gemm_bf16_nt(const bf16_t* __restrict__ A,
                                                    const bf16_t* __restrict__ W,
                                                    float* __restrict__ out,
                                                    int M, int N, int K) {
    const int lane = threadIdx.x & 31;
    const int wave = threadIdx.x >> 5;
    const int wm = wave >> 2;                 // 0..1
    const int wn = wave & 3;                  // 0..3
    const int Mbase = blockIdx.y * 64 + wm * 32;
    const int Nbase = blockIdx.x * 256 + wn * 64;
    const int lh = lane >> 4;                 // lane half (K-split select)
    const int ll = lane & 15;                 // row / col within tile

    const bf16_t* Ar0 = A + (size_t)(Mbase + ll) * K;
    const bf16_t* Ar1 = A + (size_t)(Mbase + 16 + ll) * K;
    const bf16_t* Wcp[4];
#pragma unroll
    for (int nt = 0; nt < 4; nt++)
        Wcp[nt] = W + (size_t)(Nbase + nt * 16 + ll) * K;

    v8f acc[2][4];
#pragma unroll
    for (int mt = 0; mt < 2; mt++)
#pragma unroll
        for (int nt = 0; nt < 4; nt++) acc[mt][nt] = (v8f){};

    for (int kb = 0; kb < K; kb += 32) {
        __builtin_prefetch(Ar0 + kb + 128, 0, 3);
        __builtin_prefetch(Wcp[0] + kb + 128, 0, 3);
        AFrag a0, a1;
        // A fragment: lanes 0-15 hold K {kb..kb+15}, lanes 16-31
        // K {kb+16..kb+31} -> two aligned 16B loads per fragment.
        a0.h[0] = *(const v8bf*)(Ar0 + kb + lh * 16);
        a0.h[1] = *(const v8bf*)(Ar0 + kb + 8 + lh * 16);
        a1.h[0] = *(const v8bf*)(Ar1 + kb + lh * 16);
        a1.h[1] = *(const v8bf*)(Ar1 + kb + 8 + lh * 16);
#pragma unroll
        for (int nt = 0; nt < 4; nt++) {
            // B fragment: lane = column, 16 contiguous K per lane (32B aligned).
            v16bf b = *(const v16bf*)(Wcp[nt] + kb + lh * 16);
            acc[0][nt] = wmma_bf16(a0.v, b, acc[0][nt]);
            acc[1][nt] = wmma_bf16(a1.v, b, acc[1][nt]);
        }
    }
    // C/D layout: VGPR v -> row v (lanes 0-15) / row v+8 (lanes 16-31), col = ll.
#pragma unroll
    for (int mt = 0; mt < 2; mt++)
#pragma unroll
        for (int nt = 0; nt < 4; nt++)
#pragma unroll
            for (int v = 0; v < 8; v++) {
                const int r = Mbase + mt * 16 + v + 8 * lh;
                out[(size_t)r * N + Nbase + nt * 16 + ll] = acc[mt][nt][v];
            }
}

// ---------------- RoPE + repack (B,T,C) f32 -> (B,H,T,HD) bf16 ----------------
// grid.z: 0 = q (rope), 1 = k (rope), 2 = v (copy)
__global__ __launch_bounds__(256) void rope_pack(const float* __restrict__ qf,
                                                 const float* __restrict__ kf,
                                                 const float* __restrict__ vf,
                                                 bf16_t* __restrict__ qh,
                                                 bf16_t* __restrict__ kh,
                                                 bf16_t* __restrict__ vh) {
    const size_t npairs = (size_t)Bc * Tc * Hc * (HDc / 2);
    size_t i = (size_t)blockIdx.x * blockDim.x + threadIdx.x;
    if (i >= npairs) return;
    const int j = (int)(i & (HDc / 2 - 1));       // pair index 0..31
    size_t r = i >> 5;
    const int h = (int)(r % Hc);
    const size_t bt = r / Hc;                     // b*T + t
    const int t = (int)(bt % Tc);
    const size_t b = bt / Tc;

    const size_t src = bt * Cc + (size_t)h * HDc + 2 * j;
    const size_t dst = ((b * Hc + h) * (size_t)Tc + t) * HDc + 2 * j;

    const int mode = blockIdx.z;
    if (mode == 2) {
        vh[dst]     = (bf16_t)vf[src];
        vh[dst + 1] = (bf16_t)vf[src + 1];
        return;
    }
    const float inv = __powf(10000.0f, -(2.0f * (float)j) / (float)HDc);
    const float fr = (float)t * inv;
    const float c = __cosf(fr), s = __sinf(fr);
    const float* p = mode ? kf : qf;
    bf16_t* o = mode ? kh : qh;
    const float x1 = p[src], x2 = p[src + 1];
    o[dst]     = (bf16_t)(x1 * c - x2 * s);
    o[dst + 1] = (bf16_t)(x1 * s + x2 * c);
}

// ---------------- windowed attention: one block per (b,h,window) -------------
// LDS: Q[256][64] K[256][64] Vt[64][256] bf16, S[256][256] bf16, rsum[256] f32.
// V is async-staged row-major into the (not-yet-used) S region, then
// transposed LDS->LDS into Vt.
#define SMEM_BYTES (3 * 256 * 64 * 2 + 256 * 256 * 2 + 256 * 4) // 230400 B
__global__ __launch_bounds__(256) void attn_window(const bf16_t* __restrict__ qh,
                                                   const bf16_t* __restrict__ kh,
                                                   const bf16_t* __restrict__ vh,
                                                   float* __restrict__ yf) {
    const int w = blockIdx.x, h = blockIdx.y, b = blockIdx.z;
    const int tid = threadIdx.x;
    const int lane = tid & 31, wave = tid >> 5;
    const int lh = lane >> 4, ll = lane & 15;

    extern __shared__ char smem_raw[];
    bf16_t* Qs = (bf16_t*)smem_raw;        // 256 x 64
    bf16_t* Ks = Qs + 256 * 64;            // 256 x 64
    bf16_t* Vt = Ks + 256 * 64;            // 64  x 256 (transposed)
    bf16_t* S  = Vt + 64 * 256;            // 256 x 256
    float* rsum = (float*)(S + 256 * 256); // 256
    bf16_t* Vs = S;                        // 256 x 64 staging, aliased on S

    const size_t base = (((size_t)b * Hc + h) * Tc + (size_t)w * Wc) * HDc;
    // ---- stage Q,K,V rows async (ASYNCcnt path, no VGPR round-trip) ----
    {
        const int t = tid;                 // row 0..255
        ASYNC_LDS_ROW_128B(lds_addr32(Qs + t * HDc), qh + base + (size_t)t * HDc);
        ASYNC_LDS_ROW_128B(lds_addr32(Ks + t * HDc), kh + base + (size_t)t * HDc);
        ASYNC_LDS_ROW_128B(lds_addr32(Vs + t * HDc), vh + base + (size_t)t * HDc);
    }
    WAIT_ASYNC_0();
    __syncthreads();
    // ---- transpose V in LDS: Vs[256][64] -> Vt[64][256] ----
    {
        const int t = tid;
#pragma unroll
        for (int d = 0; d < HDc; d++) Vt[d * 256 + t] = Vs[t * HDc + d];
    }
    __syncthreads();

    // ---- S = scale * Q K^T : wave owns 32 query rows (2 M-tiles) ----
    const int m0 = wave * 32;
    AFrag aq[2][2];
#pragma unroll
    for (int mt = 0; mt < 2; mt++)
#pragma unroll
        for (int ks = 0; ks < 2; ks++) {
            const bf16_t* r = Qs + (m0 + mt * 16 + ll) * HDc + ks * 32;
            aq[mt][ks].h[0] = *(const v8bf*)(r + lh * 16);
            aq[mt][ks].h[1] = *(const v8bf*)(r + 8 + lh * 16);
        }
    const float scale = 0.125f; // 1/sqrt(HD)
    for (int nt = 0; nt < 16; nt++) {
        v8f acc0 = {}, acc1 = {};
#pragma unroll
        for (int ks = 0; ks < 2; ks++) {
            // column of K^T = contiguous row slice of K
            v16bf bk = *(const v16bf*)(Ks + (nt * 16 + ll) * HDc + ks * 32 + lh * 16);
            acc0 = wmma_bf16(aq[0][ks].v, bk, acc0);
            acc1 = wmma_bf16(aq[1][ks].v, bk, acc1);
        }
#pragma unroll
        for (int v = 0; v < 8; v++) {
            S[(m0 + v + 8 * lh) * 256 + nt * 16 + ll]      = (bf16_t)(acc0[v] * scale);
            S[(m0 + 16 + v + 8 * lh) * 256 + nt * 16 + ll] = (bf16_t)(acc1[v] * scale);
        }
    }
    __syncthreads();

    // ---- softmax: one full row per thread; keep un-normalized exp in LDS ----
    {
        bf16_t* row = S + tid * 256;
        float m = -1e30f;
        for (int j = 0; j < 256; j++) m = fmaxf(m, (float)row[j]);
        float s = 0.0f;
        for (int j = 0; j < 256; j++) {
            float e = __expf((float)row[j] - m);
            s += e;
            row[j] = (bf16_t)e;
        }
        rsum[tid] = s;
    }
    __syncthreads();

    // ---- O = P V : wave owns same 32 rows x full HD (2 M-tiles x 4 N-tiles) ----
    v8f oacc[2][4];
#pragma unroll
    for (int mt = 0; mt < 2; mt++)
#pragma unroll
        for (int nt = 0; nt < 4; nt++) oacc[mt][nt] = (v8f){};

    for (int kb = 0; kb < 256; kb += 32) {
        AFrag ap[2];
#pragma unroll
        for (int mt = 0; mt < 2; mt++) {
            const bf16_t* r = S + (m0 + mt * 16 + ll) * 256 + kb;
            ap[mt].h[0] = *(const v8bf*)(r + lh * 16);
            ap[mt].h[1] = *(const v8bf*)(r + 8 + lh * 16);
        }
#pragma unroll
        for (int nt = 0; nt < 4; nt++) {
            // column d of V = contiguous row of Vt
            v16bf bv = *(const v16bf*)(Vt + (nt * 16 + ll) * 256 + kb + lh * 16);
            oacc[0][nt] = wmma_bf16(ap[0].v, bv, oacc[0][nt]);
            oacc[1][nt] = wmma_bf16(ap[1].v, bv, oacc[1][nt]);
        }
    }
    // ---- epilogue: fold 1/rowsum, write (B,T,C) f32 ----
    float* yb = yf + (((size_t)b * Tc + (size_t)w * Wc)) * Cc + h * HDc;
#pragma unroll
    for (int mt = 0; mt < 2; mt++)
#pragma unroll
        for (int nt = 0; nt < 4; nt++)
#pragma unroll
            for (int v = 0; v < 8; v++) {
                const int r = m0 + mt * 16 + v + 8 * lh;
                const int c = nt * 16 + ll;
                yb[(size_t)r * Cc + c] = oacc[mt][nt][v] / rsum[r];
            }
}

// ---------------- host side ----------------
extern "C" void kernel_launch(void* const* d_in, const int* in_sizes, int n_in,
                              void* d_out, int out_size, void* d_ws, size_t ws_size,
                              hipStream_t stream) {
    const float* x  = (const float*)d_in[0];
    const float* wq = (const float*)d_in[1];
    const float* wk = (const float*)d_in[2];
    const float* wv = (const float*)d_in[3];
    const float* wo = (const float*)d_in[4];
    float* out = (float*)d_out;

    char* ws = (char*)d_ws;
    size_t off = 0;
    bf16_t* xb  = (bf16_t*)(ws + off); off += MTOT * Cc * sizeof(bf16_t);       // 16 MB
    bf16_t* wqb = (bf16_t*)(ws + off); off += (size_t)Cc * Cc * sizeof(bf16_t); //  2 MB
    bf16_t* wkb = (bf16_t*)(ws + off); off += (size_t)Cc * Cc * sizeof(bf16_t);
    bf16_t* wvb = (bf16_t*)(ws + off); off += (size_t)Cc * Cc * sizeof(bf16_t);
    bf16_t* wob = (bf16_t*)(ws + off); off += (size_t)Cc * Cc * sizeof(bf16_t);
    float* qf = (float*)(ws + off); off += MTOT * Cc * sizeof(float);           // 32 MB
    float* kf = (float*)(ws + off); off += MTOT * Cc * sizeof(float);
    float* vf = (float*)(ws + off); off += MTOT * Cc * sizeof(float);
    bf16_t* qh = (bf16_t*)(ws + off); off += MTOT * Cc * sizeof(bf16_t);        // 16 MB
    bf16_t* kh = (bf16_t*)(ws + off); off += MTOT * Cc * sizeof(bf16_t);
    bf16_t* vh = (bf16_t*)(ws + off); off += MTOT * Cc * sizeof(bf16_t);
    float*  yf = qf;            // reuse: q f32 no longer needed after rope_pack
    bf16_t* ybf = (bf16_t*)kf;  // reuse: k f32 no longer needed after rope_pack

    const size_t nx = MTOT * Cc;            // 8388608
    const size_t nw = (size_t)Cc * Cc;      // 1048576

    // 1) convert inputs to bf16
    cvt_f32_bf16<<<dim3((nx + 255) / 256), 256, 0, stream>>>(x, xb, nx);
    cvt_f32_bf16<<<dim3((nw + 255) / 256), 256, 0, stream>>>(wq, wqb, nw);
    cvt_f32_bf16<<<dim3((nw + 255) / 256), 256, 0, stream>>>(wk, wkb, nw);
    cvt_f32_bf16<<<dim3((nw + 255) / 256), 256, 0, stream>>>(wv, wvb, nw);
    cvt_f32_bf16<<<dim3((nw + 255) / 256), 256, 0, stream>>>(wo, wob, nw);

    // 2) QKV projections (WMMA)
    dim3 ggrid(Cc / 256, MTOT / 64);        // (4, 128)
    gemm_bf16_nt<<<ggrid, 256, 0, stream>>>(xb, wqb, qf, (int)MTOT, Cc, Cc);
    gemm_bf16_nt<<<ggrid, 256, 0, stream>>>(xb, wkb, kf, (int)MTOT, Cc, Cc);
    gemm_bf16_nt<<<ggrid, 256, 0, stream>>>(xb, wvb, vf, (int)MTOT, Cc, Cc);

    // 3) RoPE + repack to (B,H,T,HD) bf16
    const size_t npairs = (size_t)Bc * Tc * Hc * (HDc / 2);
    rope_pack<<<dim3((unsigned)((npairs + 255) / 256), 1, 3), 256, 0, stream>>>(
        qf, kf, vf, qh, kh, vh);

    // 4) windowed attention (WMMA + async LDS staging), output (B,T,C) f32
    attn_window<<<dim3(NWc, Hc, Bc), 256, SMEM_BYTES, stream>>>(qh, kh, vh, yf);

    // 5) output projection (WMMA)
    cvt_f32_bf16<<<dim3((nx + 255) / 256), 256, 0, stream>>>(yf, ybf, nx);
    gemm_bf16_nt<<<ggrid, 256, 0, stream>>>(ybf, wob, out, (int)MTOT, Cc, Cc);
}
